// Model_52561809769072
// MI455X (gfx1250) — compile-verified, matrix operand order checked by reading, not run on hardware
//
#include <hip/hip_runtime.h>
#include <hip/hip_bf16.h>

typedef __bf16 bf16_t;
typedef __attribute__((ext_vector_type(16))) __bf16 v16bf;
typedef __attribute__((ext_vector_type(8)))  __bf16 v8bf;
typedef __attribute__((ext_vector_type(8)))  float   v8f;

static __device__ __forceinline__ int imind(int a, int b) { return a < b ? a : b; }
static __device__ __forceinline__ int imaxd(int a, int b) { return a > b ? a : b; }
static inline int hmin(int a, int b) { return a < b ? a : b; }

// ---------------------------------------------------------------------------
// Generic GEMM: C[M,N] = A[M,K] @ B[K,N], f32 in/out, bf16 WMMA compute.
// Epilogue: acc = acc*row_scale[m] + bias_row[m] + bias_col[n]; optional ReLU.
// 256 threads = 8 waves in a 4(m) x 2(n) grid; wave tile = 16*WMT x 16*WNT.
// LDS double-buffered; one barrier per K-step.
// Fast path (tile fully in range, uniform branch): float4 global loads.
// Slow path: unconditional clamped loads + select (no EXEC divergence).
// NOTE: all callers guarantee lda/ldb multiples of 4 and 16B-aligned bases,
// so the float4 fast path is safe.
// ---------------------------------------------------------------------------
#define LDSTR 48   // halves per LDS row (96B: keeps v8bf loads 16B-aligned)

template<int BM, int BN, int WMT, int WNT>
__global__ void __launch_bounds__(256)
gemm_bf16_wmma(const float* __restrict__ A, const float* __restrict__ B,
               float* __restrict__ Cc, int M, int N, int K,
               int lda, int ldb, int ldc,
               const float* __restrict__ bias_row,
               const float* __restrict__ bias_col,
               const float* __restrict__ row_scale,
               int relu)
{
    __shared__ __align__(16) bf16_t As[2][BM * LDSTR];
    __shared__ __align__(16) bf16_t Bs[2][BN * LDSTR];   // B^T tile: [n][k]

    const int tid  = threadIdx.x;
    const int lane = tid & 31;
    const int wave = tid >> 5;
    const int wm   = wave & 3;   // 0..3
    const int wn   = wave >> 2;  // 0..1
    const int m0   = blockIdx.y * BM;
    const int n0   = blockIdx.x * BN;

    v8f acc[WMT][WNT];
    #pragma unroll
    for (int i = 0; i < WMT; ++i)
        #pragma unroll
        for (int j = 0; j < WNT; ++j)
            acc[i][j] = (v8f){};

    auto stageA = [&](int buf, int k0) {
        if ((m0 + BM <= M) && (k0 + 32 <= K)) {          // uniform fast path
            #pragma unroll
            for (int it = 0; it < BM / 32; ++it) {       // BM*8 float4 chunks
                int c  = tid + it * 256;
                int m  = c >> 3;
                int kq = (c & 7) * 4;
                const float4 v = *(const float4*)&A[(long long)(m0 + m) * lda + k0 + kq];
                bf16_t* d = &As[buf][m * LDSTR + kq];
                d[0] = (bf16_t)v.x; d[1] = (bf16_t)v.y;
                d[2] = (bf16_t)v.z; d[3] = (bf16_t)v.w;
            }
        } else {                                         // branch-free tail path
            #pragma unroll
            for (int it = 0; it < (BM * 32) / 256; ++it) {
                int e = tid + it * 256;
                int m = e >> 5, k = e & 31;
                int gm = m0 + m, gk = k0 + k;
                int gmc = gm < M ? gm : M - 1;
                int gkc = gk < K ? gk : K - 1;
                float v = A[(long long)gmc * lda + gkc];
                As[buf][m * LDSTR + k] = (bf16_t)((gm < M && gk < K) ? v : 0.f);
            }
        }
    };

    auto stageB = [&](int buf, int k0) {
        if ((n0 + BN <= N) && (k0 + 32 <= K)) {
            constexpr int CPR = BN / 4;                  // float4 chunks per k-row
            #pragma unroll
            for (int it = 0; it < BN / 32; ++it) {
                int c  = tid + it * 256;
                int k  = c / CPR;
                int nq = (c % CPR) * 4;
                const float4 v = *(const float4*)&B[(long long)(k0 + k) * ldb + n0 + nq];
                Bs[buf][(nq + 0) * LDSTR + k] = (bf16_t)v.x;
                Bs[buf][(nq + 1) * LDSTR + k] = (bf16_t)v.y;
                Bs[buf][(nq + 2) * LDSTR + k] = (bf16_t)v.z;
                Bs[buf][(nq + 3) * LDSTR + k] = (bf16_t)v.w;
            }
        } else {
            #pragma unroll
            for (int it = 0; it < (BN * 32) / 256; ++it) {
                int e = tid + it * 256;
                int n = e % BN;                          // BN is power of two
                int k = e / BN;
                int gn = n0 + n, gk = k0 + k;
                int gnc = gn < N ? gn : N - 1;
                int gkc = gk < K ? gk : K - 1;
                float v = B[(long long)gkc * ldb + gnc];
                Bs[buf][n * LDSTR + k] = (bf16_t)((gn < N && gk < K) ? v : 0.f);
            }
        }
    };

    auto compute = [&](int buf) {
        const int r  = lane & 15;
        const int kb = (lane >> 4) * 8;
        v16bf afr[WMT], bfr[WNT];
        #pragma unroll
        for (int mt = 0; mt < WMT; ++mt) {
            int row = (wm * WMT + mt) * 16 + r;
            v8bf lo = *(const v8bf*)&As[buf][row * LDSTR + kb];
            v8bf hi = *(const v8bf*)&As[buf][row * LDSTR + 16 + kb];
            #pragma unroll
            for (int i = 0; i < 8; ++i) { afr[mt][i] = lo[i]; afr[mt][8 + i] = hi[i]; }
        }
        #pragma unroll
        for (int nt = 0; nt < WNT; ++nt) {
            int col = (wn * WNT + nt) * 16 + r;
            v8bf lo = *(const v8bf*)&Bs[buf][col * LDSTR + kb];
            v8bf hi = *(const v8bf*)&Bs[buf][col * LDSTR + 16 + kb];
            #pragma unroll
            for (int i = 0; i < 8; ++i) { bfr[nt][i] = lo[i]; bfr[nt][8 + i] = hi[i]; }
        }
        #pragma unroll
        for (int mt = 0; mt < WMT; ++mt)
            #pragma unroll
            for (int nt = 0; nt < WNT; ++nt)
                acc[mt][nt] = __builtin_amdgcn_wmma_f32_16x16x32_bf16(
                    false, afr[mt], false, bfr[nt], (short)0, acc[mt][nt], false, false);
    };

    stageA(0, 0);
    stageB(0, 0);
    __syncthreads();
    int buf = 0;
    for (int k0 = 0; k0 < K; k0 += 32) {
        if (k0 + 32 < K) { stageA(buf ^ 1, k0 + 32); stageB(buf ^ 1, k0 + 32); }
        compute(buf);
        __syncthreads();
        buf ^= 1;
    }

    const int ncl   = lane & 15;
    const int rbase = (lane >> 4) * 8;
    #pragma unroll
    for (int mt = 0; mt < WMT; ++mt) {
        #pragma unroll
        for (int rr = 0; rr < 8; ++rr) {
            int gm = m0 + (wm * WMT + mt) * 16 + rbase + rr;
            if (gm >= M) continue;
            float rs = row_scale ? row_scale[gm] : 1.f;
            float br = bias_row ? bias_row[gm] : 0.f;
            #pragma unroll
            for (int nt = 0; nt < WNT; ++nt) {
                int gn = n0 + (wn * WNT + nt) * 16 + ncl;
                if (gn < N) {
                    float v = acc[mt][nt][rr] * rs + br + (bias_col ? bias_col[gn] : 0.f);
                    if (relu) v = fmaxf(v, 0.f);
                    Cc[(long long)gm * ldc + gn] = v;
                }
            }
        }
    }
}

// ---------------------------------------------------------------------------
// im2col conv k3 s1 p1, one image.  blockIdx.y = K-row (ci*9+ky*3+kx).
// W is a power of two; wshift = log2(W).
// ---------------------------------------------------------------------------
__global__ void im2col_k3(const float* __restrict__ in, float* __restrict__ col,
                          int H, int W, int wshift)
{
    int krow = blockIdx.y;
    int kx = krow % 3, t = krow / 3;
    int ky = t % 3, ci = t / 3;
    int P = H * W;
    const float* src = in + (long long)ci * P;
    float* dst = col + (long long)krow * P;
    for (int p = blockIdx.x * blockDim.x + threadIdx.x; p < P;
         p += gridDim.x * blockDim.x) {
        int x = p & (W - 1), y = p >> wshift;
        int iy = y + ky - 1, ix = x + kx - 1;
        float v = 0.f;
        if (iy >= 0 && iy < H && ix >= 0 && ix < W)
            v = src[(iy << wshift) + ix];
        dst[p] = v;
    }
}

// ---------------------------------------------------------------------------
// im2col ConvTranspose2d k4 s2 p1, one image; output 2H x 2W.
// blockIdx.y = K-row (ci*16+ky*4+kx).  wishift = log2(Win).
// ---------------------------------------------------------------------------
__global__ void im2col_ct4(const float* __restrict__ in, float* __restrict__ col,
                           int Hin, int Win, int wishift)
{
    int krow = blockIdx.y;
    int kx = krow & 3, ky = (krow >> 2) & 3, ci = krow >> 4;
    int Wo = Win * 2;
    int P = (Hin * 2) * Wo;
    int woshift = wishift + 1;
    const float* src = in + (long long)ci * Hin * Win;
    float* dst = col + (long long)krow * P;
    for (int p = blockIdx.x * blockDim.x + threadIdx.x; p < P;
         p += gridDim.x * blockDim.x) {
        int x = p & (Wo - 1), y = p >> woshift;
        int ty = y + 1 - ky, tx = x + 1 - kx;
        float v = 0.f;
        if (ty >= 0 && tx >= 0 && ((ty & 1) == 0) && ((tx & 1) == 0)) {
            int iy = ty >> 1, ix = tx >> 1;
            if (iy < Hin && ix < Win)
                v = src[(iy << wishift) + ix];
        }
        dst[p] = v;
    }
}

// Repack ConvT weights [Cin,Cout,4,4] -> [Cout, Cin*16]
__global__ void repack_ct_w(const float* __restrict__ w, float* __restrict__ wt,
                            int Cin, int Cout)
{
    long long total = (long long)Cin * Cout * 16;
    for (long long i = blockIdx.x * (long long)blockDim.x + threadIdx.x;
         i < total; i += (long long)gridDim.x * blockDim.x) {
        int k = (int)(i & 15);
        long long t = i >> 4;
        int co = (int)(t % Cout);
        int ci = (int)(t / Cout);
        wt[((long long)co * Cin + ci) * 16 + k] =
            w[((long long)ci * Cout + co) * 16 + k];
    }
}

// ---------------------------------------------------------------------------
// BatchNorm (training statistics): mean/var per channel over [B,H,W]
// ---------------------------------------------------------------------------
__global__ void bn_stats(const float* __restrict__ x, float* __restrict__ mean,
                         float* __restrict__ var, int C, int HW, int Bn)
{
    int c = blockIdx.x;
    float s1 = 0.f, s2 = 0.f;
    for (int n = 0; n < Bn; ++n) {
        const float* base = x + ((long long)n * C + c) * HW;
        for (int p = threadIdx.x; p < HW; p += blockDim.x) {
            float v = base[p];
            s1 += v; s2 += v * v;
        }
    }
    __shared__ float r1[256], r2[256];
    r1[threadIdx.x] = s1; r2[threadIdx.x] = s2;
    __syncthreads();
    for (int s = 128; s > 0; s >>= 1) {
        if ((int)threadIdx.x < s) {
            r1[threadIdx.x] += r1[threadIdx.x + s];
            r2[threadIdx.x] += r2[threadIdx.x + s];
        }
        __syncthreads();
    }
    if (threadIdx.x == 0) {
        float cnt = (float)Bn * (float)HW;
        float m = r1[0] / cnt;
        mean[c] = m;
        var[c]  = r2[0] / cnt - m * m;
    }
}

// In-place BN + ReLU.  blockIdx.y = n*C + c  (one image-channel plane).
__global__ void bn_apply_relu(float* __restrict__ x, const float* __restrict__ mean,
                              const float* __restrict__ var, const float* __restrict__ g,
                              const float* __restrict__ b, int C, int HW)
{
    int nc = blockIdx.y;
    int c  = nc % C;
    float m  = mean[c];
    float sc = rsqrtf(var[c] + 1e-5f) * g[c];
    float bb = b[c];
    float* p = x + (long long)nc * HW;
    for (int i = blockIdx.x * blockDim.x + threadIdx.x; i < HW;
         i += gridDim.x * blockDim.x) {
        float v = (p[i] - m) * sc + bb;
        p[i] = v > 0.f ? v : 0.f;
    }
}

// ---------------------------------------------------------------------------
// GCN support kernels
// ---------------------------------------------------------------------------
// x: [5,C,h,w] (one group). feat[node*C+c], featT[c*N+node], node=(g5*h+y)*w+x
__global__ void extract_feat(const float* __restrict__ x, float* __restrict__ feat,
                             float* __restrict__ featT, int C, int h, int w)
{
    long long total = 5ll * C * h * w;
    int N = 5 * h * w;
    for (long long i = blockIdx.x * (long long)blockDim.x + threadIdx.x;
         i < total; i += (long long)gridDim.x * blockDim.x) {
        int xx = (int)(i % w);
        long long t = i / w;
        int yy = (int)(t % h); t /= h;
        int c  = (int)(t % C);
        int g5 = (int)(t / C);
        float v = x[i];
        int node = (g5 * h + yy) * w + xx;
        feat[(long long)node * C + c] = v;
        featT[(long long)c * N + node] = v;
    }
}

// s[c] = sum over nodes of featT row c (contiguous)
__global__ void colsum_kernel(const float* __restrict__ featT, float* __restrict__ s, int N)
{
    int c = blockIdx.x;
    const float* rowp = featT + (long long)c * N;
    float acc = 0.f;
    for (int i = threadIdx.x; i < N; i += blockDim.x) acc += rowp[i];
    __shared__ float red[256];
    red[threadIdx.x] = acc;
    __syncthreads();
    for (int st = 128; st > 0; st >>= 1) {
        if ((int)threadIdx.x < st) red[threadIdx.x] += red[threadIdx.x + st];
        __syncthreads();
    }
    if (threadIdx.x == 0) s[c] = red[0];
}

// inv[i] = 1/(feat[i,:]·s + 1e-10)   (rowsum of F F^T via F @ colsum(F))
__global__ void rowsum_inv(const float* __restrict__ feat, const float* __restrict__ s,
                           float* __restrict__ inv, int N, int C)
{
    int i = blockIdx.x * blockDim.x + threadIdx.x;
    if (i >= N) return;
    const float* r = feat + (long long)i * C;
    float acc = 0.f;
    for (int c = 0; c < C; ++c) acc += r[c] * s[c];
    inv[i] = 1.f / (acc + 1e-10f);
}

// Bilinear upsample GCN output [N,C] (node-major) -> gcx[group*5+g5, ch_off+c, 32,32]
__global__ void upsample_gc(const float* __restrict__ g2, float* __restrict__ gcx,
                            int C, int h, int w, int group, int ch_off)
{
    long long total = 5ll * C * 32 * 32;
    float sh = (float)h / 32.f, sw = (float)w / 32.f;
    for (long long i = blockIdx.x * (long long)blockDim.x + threadIdx.x;
         i < total; i += (long long)gridDim.x * blockDim.x) {
        int x  = (int)(i & 31);
        int y  = (int)((i >> 5) & 31);
        int c  = (int)((i >> 10) % C);
        int g5 = (int)(i / ((long long)C << 10));
        float syf = (y + 0.5f) * sh - 0.5f;
        float sxf = (x + 0.5f) * sw - 0.5f;
        int y0 = (int)floorf(syf), x0 = (int)floorf(sxf);
        float fy = syf - (float)y0, fx = sxf - (float)x0;
        int y0c = imind(imaxd(y0, 0), h - 1);
        int y1c = imind(imaxd(y0 + 1, 0), h - 1);
        int x0c = imind(imaxd(x0, 0), w - 1);
        int x1c = imind(imaxd(x0 + 1, 0), w - 1);
        const float* base = g2 + (long long)g5 * h * w * C + c;
        float v00 = base[((long long)y0c * w + x0c) * C];
        float v01 = base[((long long)y0c * w + x1c) * C];
        float v10 = base[((long long)y1c * w + x0c) * C];
        float v11 = base[((long long)y1c * w + x1c) * C];
        float v = v00 * (1 - fy) * (1 - fx) + v01 * (1 - fy) * fx
                + v10 * fy * (1 - fx) + v11 * fy * fx;
        int bimg = group * 5 + g5;
        gcx[((long long)bimg * 336 + (ch_off + c)) * 1024 + y * 32 + x] = v;
    }
}

// Final: bilinear 128->256 + sigmoid. in [20,1,128,128], out [20,256,256]
__global__ void upsample_sigmoid(const float* __restrict__ in, float* __restrict__ out)
{
    const int Hin = 128, Ho = 256;
    long long total = 20ll * Ho * Ho;
    for (long long i = blockIdx.x * (long long)blockDim.x + threadIdx.x;
         i < total; i += (long long)gridDim.x * blockDim.x) {
        int x = (int)(i & 255);
        int y = (int)((i >> 8) & 255);
        int n = (int)(i >> 16);
        float syf = (y + 0.5f) * 0.5f - 0.5f;
        float sxf = (x + 0.5f) * 0.5f - 0.5f;
        int y0 = (int)floorf(syf), x0 = (int)floorf(sxf);
        float fy = syf - (float)y0, fx = sxf - (float)x0;
        int y0c = imind(imaxd(y0, 0), Hin - 1);
        int y1c = imind(imaxd(y0 + 1, 0), Hin - 1);
        int x0c = imind(imaxd(x0, 0), Hin - 1);
        int x1c = imind(imaxd(x0 + 1, 0), Hin - 1);
        const float* base = in + (long long)n * Hin * Hin;
        float v00 = base[y0c * Hin + x0c];
        float v01 = base[y0c * Hin + x1c];
        float v10 = base[y1c * Hin + x0c];
        float v11 = base[y1c * Hin + x1c];
        float v = v00 * (1 - fy) * (1 - fx) + v01 * (1 - fy) * fx
                + v10 * fy * (1 - fx) + v11 * fy * fx;
        out[i] = 1.f / (1.f + expf(-v));
    }
}

// ---------------------------------------------------------------------------
// Host side
// ---------------------------------------------------------------------------
static inline int gs_blocks(long long total)
{
    long long b = (total + 255) / 256;
    if (b > 2048) b = 2048;
    if (b < 1) b = 1;
    return (int)b;
}

static inline int ilog2(int v) { int s = 0; while ((1 << s) < v) ++s; return s; }

static void launch_gemm(const float* A, const float* B, float* C,
                        int M, int N, int K, int lda, int ldb, int ldc,
                        const float* bias_row, const float* bias_col,
                        const float* row_scale, int relu, hipStream_t stream)
{
    if (M >= 96 && N >= 96) {
        dim3 grid((N + 127) / 128, (M + 127) / 128);
        gemm_bf16_wmma<128, 128, 2, 4><<<grid, dim3(256), 0, stream>>>(
            A, B, C, M, N, K, lda, ldb, ldc, bias_row, bias_col, row_scale, relu);
    } else if (N >= 96) {
        dim3 grid((N + 127) / 128, (M + 63) / 64);
        gemm_bf16_wmma<64, 128, 1, 4><<<grid, dim3(256), 0, stream>>>(
            A, B, C, M, N, K, lda, ldb, ldc, bias_row, bias_col, row_scale, relu);
    } else {
        dim3 grid((N + 63) / 64, (M + 63) / 64);
        gemm_bf16_wmma<64, 64, 1, 2><<<grid, dim3(256), 0, stream>>>(
            A, B, C, M, N, K, lda, ldb, ldc, bias_row, bias_col, row_scale, relu);
    }
}

static void conv3_layer(const float* in, float* out, const float* w, const float* b,
                        int Cin, int Cout, int H, int W, float* col, hipStream_t stream)
{
    int HW = H * W, K = Cin * 9;
    int ws = ilog2(W);
    dim3 ig(hmin((HW + 255) / 256, 64), Cin * 9);
    for (int n = 0; n < 20; ++n) {
        im2col_k3<<<ig, dim3(256), 0, stream>>>(in + (long long)n * Cin * HW, col, H, W, ws);
        launch_gemm(w, col, out + (long long)n * Cout * HW,
                    Cout, HW, K, K, HW, HW, b, nullptr, nullptr, 0, stream);
    }
}

static void convT_layer(const float* in, float* out, const float* wrep, const float* b,
                        int Cin, int Cout, int Hin, int Win, float* col, hipStream_t stream)
{
    int HW = 4 * Hin * Win, K = Cin * 16;
    int ws = ilog2(Win);
    dim3 ig(hmin((HW + 255) / 256, 64), Cin * 16);
    for (int n = 0; n < 20; ++n) {
        im2col_ct4<<<ig, dim3(256), 0, stream>>>(in + (long long)n * Cin * Hin * Win,
                                                 col, Hin, Win, ws);
        launch_gemm(wrep, col, out + (long long)n * Cout * HW,
                    Cout, HW, K, K, HW, HW, b, nullptr, nullptr, 0, stream);
    }
}

static void bn_relu_ip(float* x, const float* g, const float* beta, int C, int HW,
                       float* meanb, float* varb, hipStream_t stream)
{
    bn_stats<<<C, 256, 0, stream>>>(x, meanb, varb, C, HW, 20);
    dim3 grid(hmin((HW + 255) / 256, 64), 20 * C);
    bn_apply_relu<<<grid, dim3(256), 0, stream>>>(x, meanb, varb, g, beta, C, HW);
}

// One GCN branch for one group.  adj@X computed as F(F^T X)/rowsum (exact algebra).
static void gcn_scale(const float* xg, int C, int C1, int h, int w,
                      const float* w1, const float* b1,
                      const float* w2, const float* b2,
                      int group, int ch_off,
                      float* feat, float* featT, float* Y, float* G1, float* G2,
                      float* Z, float* S, float* INV, float* gcx, hipStream_t stream)
{
    int N = 5 * h * w;
    extract_feat<<<gs_blocks(5ll * C * h * w), 256, 0, stream>>>(xg, feat, featT, C, h, w);
    colsum_kernel<<<C, 256, 0, stream>>>(featT, S, N);
    rowsum_inv<<<(N + 255) / 256, 256, 0, stream>>>(feat, S, INV, N, C);
    // Layer 1: G1 = relu( F (F^T (F w1)) * inv + b1 )
    launch_gemm(feat,  w1, Y,  N, C1, C,  C,  C1, C1, nullptr, nullptr, nullptr, 0, stream);
    launch_gemm(featT, Y,  Z,  C, C1, N,  N,  C1, C1, nullptr, nullptr, nullptr, 0, stream);
    launch_gemm(feat,  Z,  G1, N, C1, C,  C,  C1, C1, nullptr, b1, INV, 1, stream);
    // Layer 2: G2 = relu( F (F^T (G1 w2)) * inv + b2 )
    launch_gemm(G1,    w2, Y,  N, C1, C1, C1, C1, C1, nullptr, nullptr, nullptr, 0, stream);
    launch_gemm(featT, Y,  Z,  C, C1, N,  N,  C1, C1, nullptr, nullptr, nullptr, 0, stream);
    launch_gemm(feat,  Z,  G2, N, C1, C,  C,  C1, C1, nullptr, b2, INV, 1, stream);
    upsample_gc<<<gs_blocks(5ll * C1 * 1024), 256, 0, stream>>>(G2, gcx, C1, h, w, group, ch_off);
}

extern "C" void kernel_launch(void* const* d_in, const int* in_sizes, int n_in,
                              void* d_out, int out_size, void* d_ws, size_t ws_size,
                              hipStream_t stream)
{
    (void)in_sizes; (void)n_in; (void)out_size; (void)ws_size;
    const float* fea2 = (const float*)d_in[0];   // [20, 96,32,32]
    const float* fea3 = (const float*)d_in[1];   // [20,192,16,16]
    const float* fea4 = (const float*)d_in[2];   // [20,384, 8, 8]
    // d_in[3] = img (only its 256x256 shape matters)
    const float* w41 = (const float*)d_in[4];  const float* b41 = (const float*)d_in[5];
    const float* w42 = (const float*)d_in[6];  const float* b42 = (const float*)d_in[7];
    const float* w31 = (const float*)d_in[8];  const float* b31 = (const float*)d_in[9];
    const float* w32 = (const float*)d_in[10]; const float* b32 = (const float*)d_in[11];
    const float* w21 = (const float*)d_in[12]; const float* b21 = (const float*)d_in[13];
    const float* w22 = (const float*)d_in[14]; const float* b22 = (const float*)d_in[15];
    const float* dec[30];
    for (int i = 0; i < 30; ++i) dec[i] = (const float*)d_in[16 + i];

    // ---- workspace carve-up ----
    char* ws = (char*)d_ws;
    size_t off = 0;
    auto carve = [&](size_t bytes) -> float* {
        float* p = (float*)(ws + off);
        off += (bytes + 255) & ~(size_t)255;
        return p;
    };
    float* gcx   = carve(20ull * 336 * 1024 * 4);        // 27.5 MB
    float* act0  = carve(20ull * 64 * 128 * 128 * 4);    // 83.9 MB
    float* act1  = carve(20ull * 64 * 128 * 128 * 4);    // 83.9 MB
    float* col   = carve(2048ull * 16384 * 4);           // 134 MB (max im2col)
    float* feat  = carve(491520ull * 4);
    float* featT = carve(491520ull * 4);
    float* Ybuf  = carve(245760ull * 4);
    float* G1buf = carve(245760ull * 4);
    float* G2buf = carve(245760ull * 4);
    float* Zbuf  = carve(73728ull * 4);
    float* Sbuf  = carve(1536);
    float* INVb  = carve(5120ull * 4);
    float* meanb = carve(1536);
    float* varb  = carve(1536);
    float* wrep  = carve(336ull * 128 * 16 * 4);         // ConvT repack buffer

    // ---- passthrough outputs: (out, fea4, fea3, fea2) ----
    float* out_map = (float*)d_out;                       // 20*256*256
    float* out_f4  = out_map + 20ll * 256 * 256;
    float* out_f3  = out_f4 + 20ll * 384 * 64;
    float* out_f2  = out_f3 + 20ll * 192 * 256;
    (void)hipMemcpyAsync(out_f4, fea4, 20ll * 384 * 64 * 4,  hipMemcpyDeviceToDevice, stream);
    (void)hipMemcpyAsync(out_f3, fea3, 20ll * 192 * 256 * 4, hipMemcpyDeviceToDevice, stream);
    (void)hipMemcpyAsync(out_f2, fea2, 20ll * 96 * 1024 * 4, hipMemcpyDeviceToDevice, stream);

    // ---- GCN branches: 4 groups x 3 scales -> gcx [20,336,32,32] ----
    for (int g = 0; g < 4; ++g) {
        gcn_scale(fea4 + (long long)g * 5 * 384 * 64,  384, 192,  8,  8,
                  w41, b41, w42, b42, g, 0,
                  feat, featT, Ybuf, G1buf, G2buf, Zbuf, Sbuf, INVb, gcx, stream);
        gcn_scale(fea3 + (long long)g * 5 * 192 * 256, 192,  96, 16, 16,
                  w31, b31, w32, b32, g, 192,
                  feat, featT, Ybuf, G1buf, G2buf, Zbuf, Sbuf, INVb, gcx, stream);
        gcn_scale(fea2 + (long long)g * 5 * 96 * 1024,  96,  48, 32, 32,
                  w21, b21, w22, b22, g, 288,
                  feat, featT, Ybuf, G1buf, G2buf, Zbuf, Sbuf, INVb, gcx, stream);
    }

    // ---- decoder ----
    conv3_layer(gcx, act0, dec[0], dec[1], 336, 336, 32, 32, col, stream);
    bn_relu_ip(act0, dec[2], dec[3], 336, 32 * 32, meanb, varb, stream);

    repack_ct_w<<<gs_blocks(336ll * 128 * 16), 256, 0, stream>>>(dec[4], wrep, 336, 128);
    convT_layer(act0, act1, wrep, dec[5], 336, 128, 32, 32, col, stream);
    bn_relu_ip(act1, dec[6], dec[7], 128, 64 * 64, meanb, varb, stream);

    conv3_layer(act1, act0, dec[8], dec[9], 128, 128, 64, 64, col, stream);
    bn_relu_ip(act0, dec[10], dec[11], 128, 64 * 64, meanb, varb, stream);

    conv3_layer(act0, act1, dec[12], dec[13], 128, 128, 64, 64, col, stream);
    bn_relu_ip(act1, dec[14], dec[15], 128, 64 * 64, meanb, varb, stream);

    repack_ct_w<<<gs_blocks(128ll * 64 * 16), 256, 0, stream>>>(dec[16], wrep, 128, 64);
    convT_layer(act1, act0, wrep, dec[17], 128, 64, 64, 64, col, stream);
    bn_relu_ip(act0, dec[18], dec[19], 64, 128 * 128, meanb, varb, stream);

    conv3_layer(act0, act1, dec[20], dec[21], 64, 64, 128, 128, col, stream);
    bn_relu_ip(act1, dec[22], dec[23], 64, 128 * 128, meanb, varb, stream);

    conv3_layer(act1, act0, dec[24], dec[25], 64, 64, 128, 128, col, stream);
    bn_relu_ip(act0, dec[26], dec[27], 64, 128 * 128, meanb, varb, stream);

    conv3_layer(act0, act1, dec[28], dec[29], 64, 1, 128, 128, col, stream);

    // ---- final bilinear 128->256 + sigmoid ----
    upsample_sigmoid<<<gs_blocks(20ll * 256 * 256), 256, 0, stream>>>(act1, out_map);
}
